// PFGAT_10230612099617
// MI455X (gfx1250) — matrix operand order repaired and checked by reading, not compiled, not run on hardware
//
#include <hip/hip_runtime.h>
#include <hip/hip_bf16.h>

// ---------------------------------------------------------------------------
// PFGAT for MI455X (gfx1250, wave32).  All big GEMMs use v_wmma_f32_16x16x32_f16.
// Pipeline exploits that only the last timestep feeds the graph stage.
// Each GEMM wave computes a 16x64 strip (4 accumulators) to reuse the A
// fragment 4x and raise WMMA density per byte fetched (bandwidth-bound).
// ---------------------------------------------------------------------------

typedef __attribute__((ext_vector_type(16))) _Float16 v16h;
typedef __attribute__((ext_vector_type(8)))  _Float16 v8h;
typedef __attribute__((ext_vector_type(8)))  float    v8f;

#define BB   4
#define TT   12
#define NN   10000
#define HH   128
#define GOUT 64
#define HOR  24

// ---------------------------------------------------------------------------
// WMMA GEMM:  C(M,N) = A(M,K) * Bt(N,K)^T  (+bias) (+relu)
// A, Bt are f16 row-major; accumulate f32; optional f32 and/or f16 stores.
// One wave computes a 16(M) x 64(N) strip; K multiple of 32, M mult of 16,
// N multiple of 64.  A/B k order contiguous (16*half + s, same in both); C per CDNA5 ISA 7.12.2.
// ---------------------------------------------------------------------------
__global__ __launch_bounds__(256)
void gemm_f16_wmma(const _Float16* __restrict__ A, const _Float16* __restrict__ Bt,
                   const float* __restrict__ bias, float* __restrict__ C32,
                   _Float16* __restrict__ C16, int M, int N, int K, int relu)
{
    const int wave = blockIdx.x * (blockDim.x >> 5) + (threadIdx.x >> 5);
    const int lane = threadIdx.x & 31;
    const int nstrips = N >> 6;                   // strips of 64 columns
    const long long strips = (long long)(M >> 4) * nstrips;
    if (wave >= strips) return;                   // wave-uniform exit
    const int ns = wave % nstrips;
    const int mt = wave / nstrips;

    const int half = lane >> 4;                   // 0: lanes 0-15, 1: lanes 16-31
    const int mrow = (mt << 4) + (lane & 15);     // A row held by this lane
    const int ncol = (ns << 6) + (lane & 15);     // first B column of strip
    const _Float16* Ap = A + (size_t)mrow * K + half * 16;   // K halves 0-15 / 16-31
    const _Float16* Bp0 = Bt + (size_t)(ncol)      * K + half * 16;
    const _Float16* Bp1 = Bt + (size_t)(ncol + 16) * K + half * 16;
    const _Float16* Bp2 = Bt + (size_t)(ncol + 32) * K + half * 16;
    const _Float16* Bp3 = Bt + (size_t)(ncol + 48) * K + half * 16;

    v8f acc0 = {0.f,0.f,0.f,0.f,0.f,0.f,0.f,0.f};
    v8f acc1 = acc0, acc2 = acc0, acc3 = acc0;

    for (int k0 = 0; k0 < K; k0 += 32) {
        v8h alo = *(const v8h*)(Ap + k0);         // K = k0+half*16 + 0..7
        v8h ahi = *(const v8h*)(Ap + k0 + 8);     // K = k0+half*16 + 8..15
        v16h af;
#pragma unroll
        for (int i = 0; i < 8; ++i) { af[i] = alo[i]; af[i + 8] = ahi[i]; }

        v8h b0l = *(const v8h*)(Bp0 + k0), b0h = *(const v8h*)(Bp0 + k0 + 8);
        v8h b1l = *(const v8h*)(Bp1 + k0), b1h = *(const v8h*)(Bp1 + k0 + 8);
        v8h b2l = *(const v8h*)(Bp2 + k0), b2h = *(const v8h*)(Bp2 + k0 + 8);
        v8h b3l = *(const v8h*)(Bp3 + k0), b3h = *(const v8h*)(Bp3 + k0 + 8);
        v16h bf0, bf1, bf2, bf3;
#pragma unroll
        for (int i = 0; i < 8; ++i) {
            bf0[i] = b0l[i]; bf0[i + 8] = b0h[i];
            bf1[i] = b1l[i]; bf1[i + 8] = b1h[i];
            bf2[i] = b2l[i]; bf2[i + 8] = b2h[i];
            bf3[i] = b3l[i]; bf3[i + 8] = b3h[i];
        }
        acc0 = __builtin_amdgcn_wmma_f32_16x16x32_f16(false, af, false, bf0,
                                                      (short)0, acc0, false, false);
        acc1 = __builtin_amdgcn_wmma_f32_16x16x32_f16(false, af, false, bf1,
                                                      (short)0, acc1, false, false);
        acc2 = __builtin_amdgcn_wmma_f32_16x16x32_f16(false, af, false, bf2,
                                                      (short)0, acc2, false, false);
        acc3 = __builtin_amdgcn_wmma_f32_16x16x32_f16(false, af, false, bf3,
                                                      (short)0, acc3, false, false);
    }

    const int mbase = (mt << 4) + (half << 3);    // C: VGPR r -> M = r + 8*half
    v8f* accs[4] = {&acc0, &acc1, &acc2, &acc3};
#pragma unroll
    for (int j = 0; j < 4; ++j) {
        const int col = ncol + 16 * j;
        const float bv = bias ? bias[col] : 0.f;
#pragma unroll
        for (int r = 0; r < 8; ++r) {
            float v = (*accs[j])[r] + bv;
            if (relu) v = fmaxf(v, 0.f);
            size_t idx = (size_t)(mbase + r) * N + col;
            if (C32) C32[idx] = v;
            if (C16) C16[idx] = (_Float16)v;
        }
    }
}

// -------- input pack: (B,T,N,8)+(B,T,N,8) -> rows (b,n,t) x 32 cols (zero pad)
__global__ __launch_bounds__(256)
void pack_input(const float* __restrict__ xh, const float* __restrict__ xc,
                _Float16* __restrict__ X, long long total)
{
    long long tid = (long long)blockIdx.x * blockDim.x + threadIdx.x;
    if (tid >= total) return;
    int c = (int)(tid & 31);
    long long r = tid >> 5;                 // r = (b*NN+n)*TT + t
    int t  = (int)(r % TT);
    long long bn = r / TT;
    int n  = (int)(bn % NN);
    int b  = (int)(bn / NN);
    float v = 0.f;
    size_t base = (((size_t)b * TT + t) * NN + n) * 8;
    if (c < 8)       v = xh[base + c];
    else if (c < 16) v = xc[base + (c - 8)];
    X[tid] = (_Float16)v;
}

// -------- weight transpose+convert: W(K,N) f32 -> Wt(N,Kpad) f16 (zero pad K)
__global__ __launch_bounds__(256)
void transpose_w(const float* __restrict__ W, _Float16* __restrict__ Wt,
                 int K, int N, int Kpad)
{
    long long tid = (long long)blockIdx.x * blockDim.x + threadIdx.x;
    if (tid >= (long long)N * Kpad) return;
    int k = (int)(tid % Kpad);
    int n = (int)(tid / Kpad);
    Wt[tid] = (k < K) ? (_Float16)W[(size_t)k * N + n] : (_Float16)0.f;
}

// -------- gather last timestep rows (f32 + f16)
__global__ __launch_bounds__(256)
void gather_last(const float* __restrict__ h32, const _Float16* __restrict__ h16,
                 float* __restrict__ o32, _Float16* __restrict__ o16, long long total)
{
    long long tid = (long long)blockIdx.x * blockDim.x + threadIdx.x;
    if (tid >= total) return;
    int c = (int)(tid & (HH - 1));
    long long m = tid >> 7;
    size_t src = ((size_t)m * TT + (TT - 1)) * HH + c;
    o32[tid] = h32[src];
    o16[tid] = h16[src];
}

// -------- temporal attention, query = last timestep only; wave per (node,head)
__global__ __launch_bounds__(256)
void attn_last(const _Float16* __restrict__ q16, const _Float16* __restrict__ kv16,
               _Float16* __restrict__ ao16, int M2)
{
    int wave = blockIdx.x * (blockDim.x >> 5) + (threadIdx.x >> 5);
    int lane = threadIdx.x & 31;
    if (wave >= M2 * 4) return;
    int m = wave >> 2, hh = wave & 3;
    float qd = (float)q16[(size_t)m * HH + hh * 32 + lane];
    float lg[TT];
    float mx = -1e30f;
#pragma unroll
    for (int t = 0; t < TT; ++t) {
        float kd = (float)kv16[((size_t)m * TT + t) * 256 + hh * 32 + lane];
        float p = qd * kd;
        for (int off = 16; off > 0; off >>= 1) p += __shfl_xor(p, off, 32);
        p *= 0.17677669529663689f;              // 1/sqrt(32)
        lg[t] = p;
        mx = fmaxf(mx, p);
    }
    float s = 0.f;
#pragma unroll
    for (int t = 0; t < TT; ++t) { lg[t] = __expf(lg[t] - mx); s += lg[t]; }
    float inv = 1.f / s;
    float o = 0.f;
#pragma unroll
    for (int t = 0; t < TT; ++t)
        o += lg[t] * inv * (float)kv16[((size_t)m * TT + t) * 256 + 128 + hh * 32 + lane];
    ao16[(size_t)m * HH + hh * 32 + lane] = (_Float16)o;
}

// -------- LayerNorm(a + b) over 128 features, one wave per row
__global__ __launch_bounds__(256)
void layernorm_res(const float* __restrict__ a, const float* __restrict__ b,
                   const float* __restrict__ g, const float* __restrict__ beta,
                   float* __restrict__ o32, _Float16* __restrict__ o16, int M)
{
    int wave = blockIdx.x * (blockDim.x >> 5) + (threadIdx.x >> 5);
    int lane = threadIdx.x & 31;
    if (wave >= M) return;
    size_t base = (size_t)wave * HH;
    float v[4];
    float s = 0.f;
#pragma unroll
    for (int j = 0; j < 4; ++j) {
        int c = j * 32 + lane;
        v[j] = a[base + c] + b[base + c];
        s += v[j];
    }
    for (int off = 16; off > 0; off >>= 1) s += __shfl_xor(s, off, 32);
    float mean = s * (1.f / 128.f);
    float q = 0.f;
#pragma unroll
    for (int j = 0; j < 4; ++j) { float d = v[j] - mean; q += d * d; }
    for (int off = 16; off > 0; off >>= 1) q += __shfl_xor(q, off, 32);
    float r = rsqrtf(q * (1.f / 128.f) + 1e-5f);
#pragma unroll
    for (int j = 0; j < 4; ++j) {
        int c = j * 32 + lane;
        float ov = (v[j] - mean) * r * g[c] + beta[c];
        o32[base + c] = ov;
        o16[base + c] = (_Float16)ov;
    }
}

// -------- GAT per-node attention coefficients s,d
__global__ __launch_bounds__(256)
void gat_coeffs(const float* __restrict__ xl, const float* __restrict__ asrc,
                const float* __restrict__ adst, float* __restrict__ s,
                float* __restrict__ d, int M2, int H, int O)
{
    long long tid = (long long)blockIdx.x * blockDim.x + threadIdx.x;
    if (tid >= (long long)M2 * H) return;
    int h = (int)(tid % H);
    const float* x = xl + (size_t)tid * O;
    float ss = 0.f, dd = 0.f;
    for (int o = 0; o < O; ++o) {
        float xv = x[o];
        ss += xv * asrc[h * O + o];
        dd += xv * adst[h * O + o];
    }
    s[tid] = ss;
    d[tid] = dd;
}

// order-preserving float <-> uint encoding for atomic max over signed floats
__device__ __forceinline__ unsigned enc_f(float f) {
    unsigned u = __float_as_uint(f);
    return (u & 0x80000000u) ? ~u : (u | 0x80000000u);
}
__device__ __forceinline__ float dec_f(unsigned e) {
    unsigned u = (e & 0x80000000u) ? (e & 0x7FFFFFFFu) : ~e;
    return __uint_as_float(u);
}

__device__ __forceinline__ void edge_sd(long long e, const int* __restrict__ ei,
                                        int E, long long BE, int* src, int* dst)
{
    if (e < BE) {
        int b = (int)(e / E);
        int j = (int)(e - (long long)b * E);
        *src = ei[j]     + b * NN;
        *dst = ei[E + j] + b * NN;
    } else {
        *src = *dst = (int)(e - BE);            // self loop
    }
}

// -------- edge pass 1: segment max (encoded atomicMax per dst,head)
__global__ __launch_bounds__(256)
void gat_edge_max(const int* __restrict__ ei, int E, long long BE, long long NE,
                  const float* __restrict__ s, const float* __restrict__ d,
                  unsigned* __restrict__ menc, int H)
{
    long long tid = (long long)blockIdx.x * blockDim.x + threadIdx.x;
    if (tid >= NE * H) return;
    int h = (int)(tid % H);
    long long e = tid / H;
    int src, dst;
    edge_sd(e, ei, E, BE, &src, &dst);
    float v = s[(size_t)src * H + h] + d[(size_t)dst * H + h];
    v = (v > 0.f) ? v : 0.2f * v;               // leaky_relu 0.2
    atomicMax(&menc[(size_t)dst * H + h], enc_f(v));
}

// -------- edge pass 2: numerator scatter-add + denominator
__global__ __launch_bounds__(256)
void gat_edge_accum(const int* __restrict__ ei, int E, long long BE, long long NE,
                    const float* __restrict__ s, const float* __restrict__ d,
                    const unsigned* __restrict__ menc, const float* __restrict__ xl,
                    float* __restrict__ num, float* __restrict__ den, int H, int O)
{
    long long tid = (long long)blockIdx.x * blockDim.x + threadIdx.x;
    if (tid >= NE * (long long)H * O) return;
    int o = (int)(tid % O);
    long long r = tid / O;
    int h = (int)(r % H);
    long long e = r / H;
    int src, dst;
    edge_sd(e, ei, E, BE, &src, &dst);
    float v = s[(size_t)src * H + h] + d[(size_t)dst * H + h];
    v = (v > 0.f) ? v : 0.2f * v;
    float ee = __expf(v - dec_f(menc[(size_t)dst * H + h]));
    if (o == 0) atomicAdd(&den[(size_t)dst * H + h], ee);
    atomicAdd(&num[((size_t)dst * H + h) * O + o],
              ee * xl[((size_t)src * H + h) * O + o]);
}

// -------- node finalize: divide, bias, optional ELU
__global__ __launch_bounds__(256)
void gat_finalize(const float* __restrict__ num, const float* __restrict__ den,
                  const float* __restrict__ bias, float* __restrict__ o32,
                  _Float16* __restrict__ o16, int M2, int H, int O, int elu)
{
    long long tid = (long long)blockIdx.x * blockDim.x + threadIdx.x;
    long long HO = (long long)H * O;
    if (tid >= (long long)M2 * HO) return;
    int c = (int)(tid % HO);
    long long m = tid / HO;
    float v = num[tid] / (den[m * H + c / O] + 1e-16f) + bias[c];
    if (elu) v = (v > 0.f) ? v : (__expf(v) - 1.f);
    if (o32) o32[tid] = v;
    if (o16) o16[tid] = (_Float16)v;
}

// -------- final projection (40000,64)@(64,24) + transpose to (B,HOR,N)
__global__ __launch_bounds__(256)
void out_proj(const float* __restrict__ g2, const float* __restrict__ W,
              const float* __restrict__ b, float* __restrict__ y)
{
    long long tid = (long long)blockIdx.x * blockDim.x + threadIdx.x;
    if (tid >= (long long)BB * HOR * NN) return;
    int n = (int)(tid % NN);
    long long r = tid / NN;
    int t  = (int)(r % HOR);
    int bb = (int)(r / HOR);
    const float* gg = g2 + ((size_t)bb * NN + n) * GOUT;
    float acc = b[t];
    for (int k = 0; k < GOUT; ++k) acc += gg[k] * W[k * HOR + t];
    y[tid] = acc;
}

// ---------------------------------------------------------------------------
extern "C" void kernel_launch(void* const* d_in, const int* in_sizes, int n_in,
                              void* d_out, int out_size, void* d_ws, size_t ws_size,
                              hipStream_t stream)
{
    const float* x_hist = (const float*)d_in[0];
    const float* x_cov  = (const float*)d_in[1];
    const int*   eidx   = (const int*)d_in[2];
    const float* enc_W  = (const float*)d_in[3];
    const float* enc_b  = (const float*)d_in[4];
    const float* Wq     = (const float*)d_in[5];
    const float* Wk     = (const float*)d_in[6];
    const float* Wv     = (const float*)d_in[7];
    const float* Wo     = (const float*)d_in[8];
    const float* ln1g   = (const float*)d_in[9];
    const float* ln1b   = (const float*)d_in[10];
    const float* W1     = (const float*)d_in[11];
    const float* b1     = (const float*)d_in[12];
    const float* W2     = (const float*)d_in[13];
    const float* b2     = (const float*)d_in[14];
    const float* ln2g   = (const float*)d_in[15];
    const float* ln2b   = (const float*)d_in[16];
    const float* g1W    = (const float*)d_in[17];
    const float* g1as   = (const float*)d_in[18];
    const float* g1ad   = (const float*)d_in[19];
    const float* g1b    = (const float*)d_in[20];
    const float* g2W    = (const float*)d_in[21];
    const float* g2as   = (const float*)d_in[22];
    const float* g2ad   = (const float*)d_in[23];
    const float* g2b    = (const float*)d_in[24];
    const float* outW   = (const float*)d_in[25];
    const float* outb   = (const float*)d_in[26];
    float* y = (float*)d_out;

    const int  M2 = BB * NN;                         // 40000
    const long long R  = (long long)M2 * TT;         // 480000 rows
    const int  E  = in_sizes[2] / 2;                 // 160000
    const long long BE = (long long)BB * E;
    const long long NE = BE + M2;                    // +self loops

    // ---- workspace carve-out (256B aligned) ----
    char* base = (char*)d_ws;
    size_t off = 0;
    auto alloc = [&](size_t bytes) -> char* {
        char* p = base + off;
        off += (bytes + 255) & ~(size_t)255;
        return p;
    };
    _Float16* X16   = (_Float16*)alloc(R * 32 * 2);       // padded K=32
    _Float16* Wt0   = (_Float16*)alloc(128 * 32 * 2);
    _Float16* Wqt   = (_Float16*)alloc(128 * 128 * 2);
    _Float16* Wkvt  = (_Float16*)alloc(256 * 128 * 2);    // [Wk^T ; Wv^T]
    _Float16* Wot   = (_Float16*)alloc(128 * 128 * 2);
    _Float16* W1t   = (_Float16*)alloc(256 * 128 * 2);
    _Float16* W2t   = (_Float16*)alloc(128 * 256 * 2);
    _Float16* G1t   = (_Float16*)alloc(128 * 128 * 2);
    _Float16* G2t   = (_Float16*)alloc(64 * 128 * 2);
    float*    H32   = (float*)   alloc(R * HH * 4);
    _Float16* H16   = (_Float16*)alloc(R * HH * 2);
    _Float16* KV16  = (_Float16*)alloc(R * 256 * 2);
    float*    HL32  = (float*)   alloc((size_t)M2 * HH * 4);
    _Float16* HL16  = (_Float16*)alloc((size_t)M2 * HH * 2);
    _Float16* Q16   = (_Float16*)alloc((size_t)M2 * HH * 2);
    _Float16* AO16  = (_Float16*)alloc((size_t)M2 * HH * 2);
    float*    GO32  = (float*)   alloc((size_t)M2 * HH * 4);
    float*    H1_32 = (float*)   alloc((size_t)M2 * HH * 4);
    _Float16* H1_16 = (_Float16*)alloc((size_t)M2 * HH * 2);
    _Float16* F1_16 = (_Float16*)alloc((size_t)M2 * 256 * 2);
    float*    F2_32 = (float*)   alloc((size_t)M2 * HH * 4);
    float*    HN32  = (float*)   alloc((size_t)M2 * HH * 4);
    _Float16* HN16  = (_Float16*)alloc((size_t)M2 * HH * 2);
    float*    XL32  = (float*)   alloc((size_t)M2 * HH * 4);
    float*    S1    = (float*)   alloc((size_t)M2 * 4 * 4);
    float*    D1    = (float*)   alloc((size_t)M2 * 4 * 4);
    unsigned* MENC1 = (unsigned*)alloc((size_t)M2 * 4 * 4);
    float*    DEN1  = (float*)   alloc((size_t)M2 * 4 * 4);
    float*    NUM1  = (float*)   alloc((size_t)M2 * HH * 4);
    _Float16* G1_16 = (_Float16*)alloc((size_t)M2 * HH * 2);
    float*    XL2   = (float*)   alloc((size_t)M2 * GOUT * 4);
    float*    S2    = (float*)   alloc((size_t)M2 * 4);
    float*    D2    = (float*)   alloc((size_t)M2 * 4);
    unsigned* MENC2 = (unsigned*)alloc((size_t)M2 * 4);
    float*    DEN2  = (float*)   alloc((size_t)M2 * 4);
    float*    NUM2  = (float*)   alloc((size_t)M2 * GOUT * 4);
    float*    G2_32 = (float*)   alloc((size_t)M2 * GOUT * 4);
    (void)ws_size; (void)n_in; (void)out_size;

    auto blk = [](long long n) { return (unsigned)((n + 255) / 256); };
    auto gemm = [&](const _Float16* A, const _Float16* Bt, const float* bias,
                    float* C32, _Float16* C16, long long M, int N, int K, int relu) {
        long long strips = (M >> 4) * (long long)(N >> 6);   // 16x64 per wave
        gemm_f16_wmma<<<(unsigned)((strips + 7) / 8), 256, 0, stream>>>(
            A, Bt, bias, C32, C16, (int)M, N, K, relu);
    };

    // ---- weight prep (tiny) ----
    transpose_w<<<blk(128 * 32), 256, 0, stream>>>(enc_W, Wt0, 16, 128, 32);
    transpose_w<<<blk(128 * 128), 256, 0, stream>>>(Wq, Wqt, 128, 128, 128);
    transpose_w<<<blk(128 * 128), 256, 0, stream>>>(Wk, Wkvt, 128, 128, 128);
    transpose_w<<<blk(128 * 128), 256, 0, stream>>>(Wv, Wkvt + 128 * 128, 128, 128, 128);
    transpose_w<<<blk(128 * 128), 256, 0, stream>>>(Wo, Wot, 128, 128, 128);
    transpose_w<<<blk(256 * 128), 256, 0, stream>>>(W1, W1t, 128, 256, 128);
    transpose_w<<<blk(128 * 256), 256, 0, stream>>>(W2, W2t, 256, 128, 256);
    transpose_w<<<blk(128 * 128), 256, 0, stream>>>(g1W, G1t, 128, 128, 128);
    transpose_w<<<blk(64 * 128), 256, 0, stream>>>(g2W, G2t, 128, 64, 128);

    // ---- temporal encoder ----
    pack_input<<<blk(R * 32), 256, 0, stream>>>(x_hist, x_cov, X16, R * 32);
    gemm(X16, Wt0, enc_b, H32, H16, R, HH, 32, 0);            // h = X@Win+b
    gemm(H16, Wkvt, nullptr, nullptr, KV16, R, 256, 128, 0);  // [k|v] all t
    gather_last<<<blk((long long)M2 * HH), 256, 0, stream>>>(H32, H16, HL32, HL16,
                                                             (long long)M2 * HH);
    gemm(HL16, Wqt, nullptr, nullptr, Q16, M2, HH, 128, 0);   // q, last t only
    attn_last<<<blk((long long)M2 * 4 * 32), 256, 0, stream>>>(Q16, KV16, AO16, M2);
    gemm(AO16, Wot, nullptr, GO32, nullptr, M2, HH, 128, 0);  // ao@Wo
    layernorm_res<<<blk((long long)M2 * 32), 256, 0, stream>>>(HL32, GO32, ln1g, ln1b,
                                                               H1_32, H1_16, M2);
    gemm(H1_16, W1t, b1, nullptr, F1_16, M2, 256, 128, 1);    // relu(h1@W1+b1)
    gemm(F1_16, W2t, b2, F2_32, nullptr, M2, HH, 256, 0);     // @W2+b2
    layernorm_res<<<blk((long long)M2 * 32), 256, 0, stream>>>(H1_32, F2_32, ln2g, ln2b,
                                                               HN32, HN16, M2);

    // ---- GAT layer 1 (H=4, O=32, concat, ELU) ----
    gemm(HN16, G1t, nullptr, XL32, nullptr, M2, HH, 128, 0);
    gat_coeffs<<<blk((long long)M2 * 4), 256, 0, stream>>>(XL32, g1as, g1ad, S1, D1,
                                                           M2, 4, 32);
    hipMemsetAsync(MENC1, 0, (size_t)M2 * 4 * 4, stream);
    hipMemsetAsync(DEN1, 0, (size_t)M2 * 4 * 4, stream);
    hipMemsetAsync(NUM1, 0, (size_t)M2 * HH * 4, stream);
    gat_edge_max<<<blk(NE * 4), 256, 0, stream>>>(eidx, E, BE, NE, S1, D1, MENC1, 4);
    gat_edge_accum<<<blk(NE * 128), 256, 0, stream>>>(eidx, E, BE, NE, S1, D1, MENC1,
                                                      XL32, NUM1, DEN1, 4, 32);
    gat_finalize<<<blk((long long)M2 * HH), 256, 0, stream>>>(NUM1, DEN1, g1b,
                                                              nullptr, G1_16, M2, 4, 32, 1);

    // ---- GAT layer 2 (H=1, O=64, mean==identity) ----
    gemm(G1_16, G2t, nullptr, XL2, nullptr, M2, GOUT, 128, 0);
    gat_coeffs<<<blk((long long)M2), 256, 0, stream>>>(XL2, g2as, g2ad, S2, D2,
                                                       M2, 1, 64);
    hipMemsetAsync(MENC2, 0, (size_t)M2 * 4, stream);
    hipMemsetAsync(DEN2, 0, (size_t)M2 * 4, stream);
    hipMemsetAsync(NUM2, 0, (size_t)M2 * GOUT * 4, stream);
    gat_edge_max<<<blk(NE), 256, 0, stream>>>(eidx, E, BE, NE, S2, D2, MENC2, 1);
    gat_edge_accum<<<blk(NE * 64), 256, 0, stream>>>(eidx, E, BE, NE, S2, D2, MENC2,
                                                     XL2, NUM2, DEN2, 1, 64);
    gat_finalize<<<blk((long long)M2 * GOUT), 256, 0, stream>>>(NUM2, DEN2, g2b,
                                                                G2_32, nullptr, M2, 1, 64, 0);

    // ---- output projection + transpose ----
    out_proj<<<blk((long long)BB * HOR * NN), 256, 0, stream>>>(G2_32, outW, outb, y);
}
